// SingleHeadAttention_38079180046922
// MI455X (gfx1250) — matrix operand order repaired and checked by reading, not compiled, not run on hardware
//
#include <hip/hip_runtime.h>
#include <hip/hip_bf16.h>

// ---------- types ----------
typedef __attribute__((ext_vector_type(16))) __bf16 v16bf;
typedef __attribute__((ext_vector_type(8)))  float  v8f;

union U16 {
    unsigned short u[16];
    unsigned int   w[8];
    v16bf          v;
};

__device__ __forceinline__ unsigned short f2bf(float f) {
    unsigned int u = __float_as_uint(f);
    unsigned int r = u + 0x7FFFu + ((u >> 16) & 1u);   // round-to-nearest-even
    return (unsigned short)(r >> 16);
}

// packed f32x2 -> bf16x2 (hardware cvt via HIP header)
__device__ __forceinline__ unsigned int f2bf_pk(float lo, float hi) {
    __hip_bfloat162 h2 = __float22bfloat162_rn(make_float2(lo, hi));
    union { __hip_bfloat162 h; unsigned int u; } cvt;
    cvt.h = h2;
    return cvt.u;
}

#define WMMA_BF16(A, B, Cacc) \
    __builtin_amdgcn_wmma_f32_16x16x32_bf16(false, (A), false, (B), (short)0, (Cacc), false, false)

// Problem constants
#define BB 8
#define SS 2048
#define CC 1024
#define DD 128
#define MM (BB*SS)              // 16384 rows
static __device__ const float kScale = 0.08838834764831845f; // 1/sqrt(128)

// =====================================================================
// Kernel 1: projections  Q/K/V = X @ W   (one of three per blockIdx.y)
// 4 waves/block, each wave owns a 16-row tile, all 8 N-tiles (D=128).
// W^T tile (32k x 128n) staged in LDS as bf16 per K-step, shared.
// =====================================================================
__global__ __launch_bounds__(128) void proj_kernel(
    const float* __restrict__ X,
    const float* __restrict__ Wq, const float* __restrict__ Wk, const float* __restrict__ Wv,
    unsigned short* __restrict__ Qb, unsigned short* __restrict__ Kb,
    unsigned short* __restrict__ Vb)
{
    __shared__ unsigned short Wt[128 * 34];          // [n][k], padded stride
    const int tid  = threadIdx.x;
    const int wave = tid >> 5;
    const int lane = tid & 31;
    const int n_l  = lane & 15;
    const int half = lane >> 4;
    const int row0 = (blockIdx.x * 4 + wave) * 16;
    const int which = blockIdx.y;
    const float* W = (which == 0) ? Wq : ((which == 1) ? Wk : Wv);
    unsigned short* Out = (which == 0) ? Qb : ((which == 1) ? Kb : Vb);

    v8f acc[8];
#pragma unroll
    for (int t = 0; t < 8; ++t) acc[t] = (v8f){0.f,0.f,0.f,0.f,0.f,0.f,0.f,0.f};

    for (int k0 = 0; k0 < CC; k0 += 32) {
        __syncthreads();
        // stage W^T (bf16): 16 kk-pairs x 128 n = 2048 dword stores, 128 thr x 16
#pragma unroll
        for (int i = 0; i < 16; ++i) {
            int e   = tid + i * 128;
            int kkp = e >> 7;            // 0..15 (pair index)
            int n   = e & 127;
            float w0 = W[(size_t)(k0 + 2*kkp)     * DD + n];
            float w1 = W[(size_t)(k0 + 2*kkp + 1) * DD + n];
            *(unsigned int*)&Wt[n * 34 + 2*kkp] = f2bf_pk(w0, w1);
        }
        __syncthreads();

        // A operand: X rows row0..row0+15, K = k0..k0+31 (contiguous k = 16*half + s)
        U16 a;
#pragma unroll
        for (int v = 0; v < 8; ++v) {
            int kb = half * 16 + v * 2;
            const float* p = X + (size_t)(row0 + n_l) * CC + k0 + kb;
            a.w[v] = f2bf_pk(p[0], p[1]);
        }
#pragma unroll
        for (int nt = 0; nt < 8; ++nt) {
            U16 bm;
#pragma unroll
            for (int v = 0; v < 8; ++v)
                bm.w[v] = *(const unsigned int*)&Wt[(nt*16 + n_l) * 34 + half*16 + 2*v];
            acc[nt] = WMMA_BF16(a.v, bm.v, acc[nt]);
        }
    }

    // C/D layout: lane n = lane%16, VGPR v -> row = v + 8*half
#pragma unroll
    for (int nt = 0; nt < 8; ++nt) {
#pragma unroll
        for (int v = 0; v < 8; ++v) {
            int row = row0 + v + 8*half;
            int col = nt*16 + n_l;
            Out[(size_t)row * DD + col] = f2bf(acc[nt][v]);
        }
    }
}

// =====================================================================
// Kernel 2: per-key-column softmax stats over the QUERY axis.
// One wave per (batch, 16-key tile). Online max/sum across 128 i-tiles.
// =====================================================================
__global__ __launch_bounds__(256) void stats_kernel(
    const unsigned short* __restrict__ Qb, const unsigned short* __restrict__ Kb,
    float* __restrict__ gM, float* __restrict__ gZ)
{
    const int tid  = threadIdx.x;
    const int lane = tid & 31;
    const int gw   = blockIdx.x * 8 + (tid >> 5);
    const int b    = gw >> 7;
    const int jt   = gw & 127;
    const int n_l  = lane & 15;
    const int half = lane >> 4;
    const int j0   = jt * 16;
    const unsigned short* Qbase = Qb + (size_t)b * SS * DD;
    const unsigned short* Kbase = Kb + (size_t)b * SS * DD;

    // K rows for this tile are reused every iteration: load B operands once.
    U16 bm[4];
#pragma unroll
    for (int ks = 0; ks < 4; ++ks)
#pragma unroll
        for (int v = 0; v < 8; ++v)
            bm[ks].w[v] = *(const unsigned int*)(Kbase + (size_t)(j0 + n_l) * DD
                                                 + ks*32 + half*16 + 2*v);

    float rm = -3.0e38f, rs = 0.0f;
    for (int it = 0; it < 128; ++it) {
        int i0 = it * 16;
        // prefetch next i-tile's Q row
        if (it < 127)
            __builtin_prefetch(Qbase + (size_t)(i0 + 16 + n_l) * DD, 0, 0);
        v8f c = (v8f){0.f,0.f,0.f,0.f,0.f,0.f,0.f,0.f};
#pragma unroll
        for (int ks = 0; ks < 4; ++ks) {
            U16 a;
#pragma unroll
            for (int v = 0; v < 8; ++v) {
                int kbA = half * 16 + v * 2;
                a.w[v] = *(const unsigned int*)(Qbase + (size_t)(i0 + n_l) * DD
                                                + ks*32 + kbA);
            }
            c = WMMA_BF16(a.v, bm[ks].v, c);
        }
        float lm = -3.0e38f;
#pragma unroll
        for (int v = 0; v < 8; ++v) lm = fmaxf(lm, c[v] * kScale);
        float nm = fmaxf(rm, lm);
        float s  = rs * __expf(rm - nm);
#pragma unroll
        for (int v = 0; v < 8; ++v) s += __expf(c[v] * kScale - nm);
        rm = nm; rs = s;
    }
    // merge the two lane-halves (same column j lives in lane L and L^16)
    float om = __shfl_xor(rm, 16, 32);
    float os = __shfl_xor(rs, 16, 32);
    float nm = fmaxf(rm, om);
    float tot = rs * __expf(rm - nm) + os * __expf(om - nm);
    if (lane < 16) {
        gM[(size_t)b * SS + j0 + n_l] = nm;
        gZ[(size_t)b * SS + j0 + n_l] = 1.0f / tot;
    }
}

// =====================================================================
// Kernel 3: O = [exp(S - M_j) * invZ_j] @ V.
// 8 waves/block (wave w -> query tile (blockIdx.x&31)*4 + w), V tile (bf16) shared in LDS per 32-key
// step; invZ folded into P. P re-laid out C->A through per-wave LDS.
// =====================================================================
__global__ __launch_bounds__(256) void out_kernel(
    const unsigned short* __restrict__ Qb, const unsigned short* __restrict__ Kb,
    const unsigned short* __restrict__ Vb, const float* __restrict__ gM,
    const float* __restrict__ gZ, float* __restrict__ out)
{
    __shared__ unsigned short Vt[128 * 34];     // [d][jj] padded
    __shared__ unsigned short Pl[4][16 * 34];   // per-wave [m][kk] padded
    const int tid  = threadIdx.x;
    const int wave = tid >> 5;
    const int lane = tid & 31;
    const int n_l  = lane & 15;
    const int half = lane >> 4;
    const int b    = blockIdx.x >> 5;
    const int itile = (blockIdx.x & 31) * 4 + wave;
    const int i0   = itile * 16;
    const unsigned short* Qbase = Qb + (size_t)b * SS * DD;
    const unsigned short* Kbase = Kb + (size_t)b * SS * DD;
    const unsigned short* Vbase = Vb + (size_t)b * SS * DD;

    // Preload this tile's Q operand (A) once: rows i0..i0+15, K = 0..127
    U16 qa[4];
#pragma unroll
    for (int ks = 0; ks < 4; ++ks)
#pragma unroll
        for (int v = 0; v < 8; ++v) {
            int kbA = half * 16 + v * 2;
            qa[ks].w[v] = *(const unsigned int*)(Qbase + (size_t)(i0 + n_l) * DD
                                                 + ks*32 + kbA);
        }

    v8f acc[8];
#pragma unroll
    for (int t = 0; t < 8; ++t) acc[t] = (v8f){0.f,0.f,0.f,0.f,0.f,0.f,0.f,0.f};

    for (int j0 = 0; j0 < SS; j0 += 32) {
        __syncthreads();
        // stage V tile transposed [d][jj]: 128 d x 16 jj-pairs, 256 thr x 8
#pragma unroll
        for (int i = 0; i < 8; ++i) {
            int e   = tid + i * 256;
            int jjp = e >> 7;            // 0..15 (jj pair)
            int d   = e & 127;
            unsigned int lo = Vbase[(size_t)(j0 + 2*jjp)     * DD + d];
            unsigned int hi = Vbase[(size_t)(j0 + 2*jjp + 1) * DD + d];
            *(unsigned int*)&Vt[d * 34 + 2*jjp] = lo | (hi << 16);
        }
        if (j0 + 32 < SS)
            __builtin_prefetch(Vbase + (size_t)(j0 + 32 + (tid & 31)) * DD, 0, 0);
        __syncthreads();

        // two 16-key score sub-tiles -> P (bf16, invZ folded) in per-wave LDS
#pragma unroll
        for (int sub = 0; sub < 2; ++sub) {
            int jj0 = j0 + sub * 16;
            v8f c = (v8f){0.f,0.f,0.f,0.f,0.f,0.f,0.f,0.f};
#pragma unroll
            for (int ks = 0; ks < 4; ++ks) {
                U16 bm;
#pragma unroll
                for (int v = 0; v < 8; ++v)
                    bm.w[v] = *(const unsigned int*)(Kbase + (size_t)(jj0 + n_l) * DD
                                                     + ks*32 + half*16 + 2*v);
                c = WMMA_BF16(qa[ks].v, bm.v, c);
            }
            float Mj = gM[(size_t)b * SS + jj0 + n_l];
            float Zi = gZ[(size_t)b * SS + jj0 + n_l];
#pragma unroll
            for (int v = 0; v < 8; ++v) {
                float p = __expf(c[v] * kScale - Mj) * Zi;
                int m = v + 8 * half;
                Pl[wave][m * 34 + sub * 16 + n_l] = f2bf(p);
            }
        }

        // O-tile accumulation: A = P (16x32), B = V^T slices (32x16 per d-tile)
        U16 a;
#pragma unroll
        for (int v = 0; v < 8; ++v) {
            int kb = half * 16 + v * 2;
            a.w[v] = *(const unsigned int*)&Pl[wave][n_l * 34 + kb];
        }
#pragma unroll
        for (int dt = 0; dt < 8; ++dt) {
            U16 bm;
#pragma unroll
            for (int v = 0; v < 8; ++v)
                bm.w[v] = *(const unsigned int*)&Vt[(dt*16 + n_l) * 34 + half*16 + 2*v];
            acc[dt] = WMMA_BF16(a.v, bm.v, acc[dt]);
        }
    }

#pragma unroll
    for (int dt = 0; dt < 8; ++dt) {
#pragma unroll
        for (int v = 0; v < 8; ++v) {
            out[(size_t)b * SS * DD + (size_t)(i0 + v + 8*half) * DD + dt*16 + n_l] = acc[dt][v];
        }
    }
}

// =====================================================================
extern "C" void kernel_launch(void* const* d_in, const int* in_sizes, int n_in,
                              void* d_out, int out_size, void* d_ws, size_t ws_size,
                              hipStream_t stream) {
    (void)in_sizes; (void)n_in; (void)out_size; (void)ws_size;
    const float* X  = (const float*)d_in[0];
    const float* Wq = (const float*)d_in[1];
    const float* Wk = (const float*)d_in[2];
    const float* Wv = (const float*)d_in[3];

    char* ws = (char*)d_ws;
    unsigned short* Qb = (unsigned short*)ws;                              // 4 MB
    unsigned short* Kb = (unsigned short*)(ws + (size_t)4  * 1024 * 1024); // 4 MB
    unsigned short* Vb = (unsigned short*)(ws + (size_t)8  * 1024 * 1024); // 4 MB
    float*          gM = (float*)        (ws + (size_t)12 * 1024 * 1024); // 64 KB
    float*          gZ = gM + BB * SS;                                     // 64 KB
    float*          out = (float*)d_out;

    proj_kernel<<<dim3(MM / 64, 3), 128, 0, stream>>>(X, Wq, Wk, Wv, Qb, Kb, Vb);
    stats_kernel<<<dim3((BB * (SS / 16)) / 8), 256, 0, stream>>>(Qb, Kb, gM, gZ);
    out_kernel<<<dim3(BB * (SS / 16) / 4), 256, 0, stream>>>(Qb, Kb, Vb, gM, gZ, out);
}
